// InterClassLoss_8675833938131
// MI455X (gfx1250) — compile-verified, matrix operand order checked by reading, not run on hardware
//
#include <hip/hip_runtime.h>
#include <hip/hip_bf16.h>

typedef __attribute__((ext_vector_type(16))) __bf16 v16bf;
typedef __attribute__((ext_vector_type(8)))  float   v8f;

#define B_    64
#define C_    2048
#define D_    512
#define M_    128
#define T_    100
#define NN_   1920
#define TPAD_ 112
#define MARGIN_ 1.0f
#define EPS_    1e-6f

__device__ __forceinline__ float waveReduceSum(float v) {
#pragma unroll
    for (int m = 16; m >= 1; m >>= 1) v += __shfl_xor(v, m, 32);
    return v;
}

// ---------------------------------------------------------------------------
// Kernel 1a: gather + L2-normalize non-MUC rows -> bf16 [B, NN, D]
// one wave per row; lane reads floats at (lane + 32*j) -> fully coalesced
// ---------------------------------------------------------------------------
__global__ __launch_bounds__(128) void k_norm_non(
        const float* __restrict__ emb, const int* __restrict__ nonIdx,
        __bf16* __restrict__ nonBf) {
    int wid  = (blockIdx.x * blockDim.x + threadIdx.x) >> 5;
    int lane = threadIdx.x & 31;
    if (wid >= B_ * NN_) return;
    int b = wid / NN_, n = wid % NN_;
    const float* src = emb + ((size_t)b * C_ + nonIdx[b * NN_ + n]) * D_;
    float v[16]; float s = 0.f;
#pragma unroll
    for (int i = 0; i < 16; ++i) { v[i] = src[lane + 32 * i]; s += v[i] * v[i]; }
    s = waveReduceSum(s);
    float inv = 1.0f / fmaxf(sqrtf(s), 1e-12f);
    __bf16* dst = nonBf + (size_t)wid * D_;
#pragma unroll
    for (int i = 0; i < 16; ++i) dst[lane + 32 * i] = (__bf16)(v[i] * inv);
}

// ---------------------------------------------------------------------------
// Kernel 1b: gather + normalize anchors (muc[anchor_idx]) -> bf16 [B, TPAD, D]
// rows T..TPAD-1 zero-padded so WMMA tiles are always full
// ---------------------------------------------------------------------------
__global__ __launch_bounds__(128) void k_norm_anchor(
        const float* __restrict__ emb, const int* __restrict__ mucIdx,
        const int* __restrict__ anchorIdx, __bf16* __restrict__ aBf) {
    int wid  = (blockIdx.x * blockDim.x + threadIdx.x) >> 5;
    int lane = threadIdx.x & 31;
    if (wid >= B_ * TPAD_) return;
    int b = wid / TPAD_, t = wid % TPAD_;
    __bf16* dst = aBf + (size_t)wid * D_;
    if (t >= T_) {
#pragma unroll
        for (int i = 0; i < 16; ++i) dst[lane + 32 * i] = (__bf16)0.0f;
        return;
    }
    int row = mucIdx[b * M_ + anchorIdx[b * T_ + t]];
    const float* src = emb + ((size_t)b * C_ + row) * D_;
    float v[16]; float s = 0.f;
#pragma unroll
    for (int i = 0; i < 16; ++i) { v[i] = src[lane + 32 * i]; s += v[i] * v[i]; }
    s = waveReduceSum(s);
    float inv = 1.0f / fmaxf(sqrtf(s), 1e-12f);
#pragma unroll
    for (int i = 0; i < 16; ++i) dst[lane + 32 * i] = (__bf16)(v[i] * inv);
}

// ---------------------------------------------------------------------------
// Kernel 2: hard-negative mining. grid = (TPAD/16, B), block = 128 (4 waves).
// A tile (16 x 512 bf16) staged LDS -> registers ONCE (16 frags = 128 VGPRs);
// steady-state loop per 16x16 C tile is: 2x global_load_b128 (next B frag,
// double-buffered) + 1x v_wmma_f32_16x16x32_bf16 -- no DS traffic, waits at
// nonzero loadcnt so VMEM overlaps matrix issue within one wave.
// ---------------------------------------------------------------------------
__global__ __launch_bounds__(128) void k_mine(
        const __bf16* __restrict__ aBf, const __bf16* __restrict__ nonBf,
        int* __restrict__ negIdx) {
    __shared__ __align__(16) __bf16 aT[16 * D_];
    __shared__ float sMax[4][16];
    __shared__ int   sIdx[4][16];

    int b = blockIdx.y, rt = blockIdx.x;
    int tid = threadIdx.x, lane = tid & 31, wave = tid >> 5;
    int half = lane >> 4, ln = lane & 15;

    // stage A tile (16 contiguous rows = 16 KB) into LDS
    {
        const uint4* srcA = (const uint4*)(aBf + ((size_t)b * TPAD_ + (size_t)rt * 16) * D_);
        uint4* dstA = (uint4*)aT;
#pragma unroll
        for (int i = 0; i < 8; ++i) dstA[tid + 128 * i] = srcA[tid + 128 * i];
    }
    __syncthreads();

    union F { uint4 q[2]; v16bf v; };

    // hoist all 16 A fragments (K = 0..511) into registers: reused for every
    // column tile, eliminating per-WMMA ds_load traffic
    F afrag[16];
#pragma unroll
    for (int kk = 0; kk < 16; ++kk) {
        afrag[kk].q[0] = *(const uint4*)(&aT[ln * D_ + kk * 32 + 8 * half]);
        afrag[kk].q[1] = *(const uint4*)(&aT[ln * D_ + kk * 32 + 16 + 8 * half]);
    }

    float best[8]; int besti[8];
#pragma unroll
    for (int r = 0; r < 8; ++r) { best[r] = -2.0f; besti[r] = 0; }

#define LOADB_(dst, base, koff)                                          \
    do {                                                                 \
        (dst).q[0] = *(const uint4*)((base) + (koff) + 8 * half);        \
        (dst).q[1] = *(const uint4*)((base) + (koff) + 16 + 8 * half);   \
    } while (0)

    for (int ct = wave; ct < NN_ / 16; ct += 4) {
        int col = ct * 16 + ln;                        // this lane's N column
        const __bf16* pB = nonBf + ((size_t)b * NN_ + col) * D_;

        F bbuf[2];
        LOADB_(bbuf[0], pB, 0);                        // prime the pipeline
        v8f c = {0.f, 0.f, 0.f, 0.f, 0.f, 0.f, 0.f, 0.f};
#pragma unroll
        for (int kk = 0; kk < 16; ++kk) {
            if (kk < 15) LOADB_(bbuf[(kk + 1) & 1], pB, (kk + 1) * 32);
            c = __builtin_amdgcn_wmma_f32_16x16x32_bf16(
                    false, afrag[kk].v, false, bbuf[kk & 1].v, (short)0, c,
                    false, false);
        }
#pragma unroll
        for (int r = 0; r < 8; ++r) {
            float cv = c[r];                 // row = r + 8*half, col = lane's col
            if (cv > best[r]) { best[r] = cv; besti[r] = col; }  // '>' keeps lowest col on tie
        }
    }
#undef LOADB_

    // butterfly reduce across the 16 lanes of each half (tie -> smaller index)
#pragma unroll
    for (int m = 1; m < 16; m <<= 1) {
#pragma unroll
        for (int r = 0; r < 8; ++r) {
            float om = __shfl_xor(best[r],  m, 32);
            int   oi = __shfl_xor(besti[r], m, 32);
            if (om > best[r] || (om == best[r] && oi < besti[r])) {
                best[r] = om; besti[r] = oi;
            }
        }
    }
    if (ln == 0) {
#pragma unroll
        for (int r = 0; r < 8; ++r) {
            sMax[wave][half * 8 + r] = best[r];
            sIdx[wave][half * 8 + r] = besti[r];
        }
    }
    __syncthreads();

    if (tid < 16) {
        float bm = sMax[0][tid]; int bi = sIdx[0][tid];
#pragma unroll
        for (int w = 1; w < 4; ++w) {
            float om = sMax[w][tid]; int oi = sIdx[w][tid];
            if (om > bm || (om == bm && oi < bi)) { bm = om; bi = oi; }
        }
        int t = rt * 16 + tid;
        if (t < T_) negIdx[b * T_ + t] = bi;
    }
}

// ---------------------------------------------------------------------------
// Kernel 3: exact f32 triplet terms. one wave per (b,t).
// ---------------------------------------------------------------------------
__global__ __launch_bounds__(128) void k_loss(
        const float* __restrict__ emb, const int* __restrict__ mucIdx,
        const int* __restrict__ nonIdx, const int* __restrict__ anchorIdx,
        const int* __restrict__ posIdx, const int* __restrict__ negIdx,
        float* __restrict__ lossBt) {
    int wid  = (blockIdx.x * blockDim.x + threadIdx.x) >> 5;
    int lane = threadIdx.x & 31;
    if (wid >= B_ * T_) return;
    int b = wid / T_, t = wid % T_;
    int ar = mucIdx[b * M_ + anchorIdx[b * T_ + t]];
    int pr = mucIdx[b * M_ + posIdx[b * T_ + t]];
    int nr = nonIdx[b * NN_ + negIdx[b * T_ + t]];
    const float* pa = emb + ((size_t)b * C_ + ar) * D_;
    const float* pp = emb + ((size_t)b * C_ + pr) * D_;
    const float* pn = emb + ((size_t)b * C_ + nr) * D_;

    float a[16], p[16], n[16];
    float sa = 0.f, sp = 0.f, sn = 0.f;
#pragma unroll
    for (int i = 0; i < 16; ++i) {
        a[i] = pa[lane + 32 * i]; sa += a[i] * a[i];
        p[i] = pp[lane + 32 * i]; sp += p[i] * p[i];
        n[i] = pn[lane + 32 * i]; sn += n[i] * n[i];
    }
    sa = waveReduceSum(sa); sp = waveReduceSum(sp); sn = waveReduceSum(sn);
    float ia = 1.0f / fmaxf(sqrtf(sa), 1e-12f);
    float ip = 1.0f / fmaxf(sqrtf(sp), 1e-12f);
    float in = 1.0f / fmaxf(sqrtf(sn), 1e-12f);

    float dp = 0.f, dn = 0.f;
#pragma unroll
    for (int i = 0; i < 16; ++i) {
        float av = a[i] * ia;
        float d1 = av - p[i] * ip + EPS_; dp += d1 * d1;
        float d2 = av - n[i] * in + EPS_; dn += d2 * d2;
    }
    dp = waveReduceSum(dp); dn = waveReduceSum(dn);
    if (lane == 0)
        lossBt[wid] = fmaxf(sqrtf(dp) - sqrtf(dn) + MARGIN_, 0.0f);
}

// ---------------------------------------------------------------------------
// Kernel 4: deterministic final reduction -> d_out[0] = sum / (B*T)
// ---------------------------------------------------------------------------
__global__ __launch_bounds__(256) void k_reduce(
        const float* __restrict__ lossBt, float* __restrict__ out) {
    __shared__ float s[256];
    float acc = 0.f;
    for (int i = threadIdx.x; i < B_ * T_; i += 256) acc += lossBt[i];
    s[threadIdx.x] = acc;
    __syncthreads();
    for (int m = 128; m > 0; m >>= 1) {
        if (threadIdx.x < m) s[threadIdx.x] += s[threadIdx.x + m];
        __syncthreads();
    }
    if (threadIdx.x == 0) out[0] = s[0] / (float)(B_ * T_);
}

// ---------------------------------------------------------------------------
extern "C" void kernel_launch(void* const* d_in, const int* in_sizes, int n_in,
                              void* d_out, int out_size, void* d_ws, size_t ws_size,
                              hipStream_t stream) {
    const float* emb       = (const float*)d_in[0];  // [B, C, D] f32
    const int*   mucIdx    = (const int*)  d_in[1];  // [B, M]
    const int*   nonIdx    = (const int*)  d_in[2];  // [B, NN]
    const int*   anchorIdx = (const int*)  d_in[3];  // [B, T]
    const int*   posIdx    = (const int*)  d_in[4];  // [B, T]
    float*       out       = (float*)d_out;

    char* ws = (char*)d_ws;
    size_t off = 0;
    __bf16* nonBf = (__bf16*)(ws + off); off += (size_t)B_ * NN_   * D_ * sizeof(__bf16);
    __bf16* aBf   = (__bf16*)(ws + off); off += (size_t)B_ * TPAD_ * D_ * sizeof(__bf16);
    int*    negI  = (int*)   (ws + off); off += (size_t)B_ * T_ * sizeof(int);
    float*  lossBt= (float*) (ws + off);
    (void)in_sizes; (void)n_in; (void)out_size; (void)ws_size;

    // 1a: normalize/gather non-MUC rows (64*1920 waves, 4 waves/block)
    k_norm_non<<<(B_ * NN_) / 4, 128, 0, stream>>>(emb, nonIdx, nonBf);
    // 1b: normalize/gather anchors (64*112 waves)
    k_norm_anchor<<<(B_ * TPAD_) / 4, 128, 0, stream>>>(emb, mucIdx, anchorIdx, aBf);
    // 2: WMMA mining
    k_mine<<<dim3(TPAD_ / 16, B_), 128, 0, stream>>>(aBf, nonBf, negI);
    // 3: exact loss terms (6400 waves)
    k_loss<<<(B_ * T_) / 4, 128, 0, stream>>>(emb, mucIdx, nonIdx, anchorIdx,
                                              posIdx, negI, lossBt);
    // 4: deterministic scalar reduction
    k_reduce<<<1, 256, 0, stream>>>(lossBt, out);
}
